// TimeMixChunkParallel_71966472012323
// MI455X (gfx1250) — matrix operand order repaired and checked by reading, not compiled, not run on hardware
//
#include <hip/hip_runtime.h>
#include <hip/hip_bf16.h>

// ---------------- constants ----------------
#define D_MODEL 2048
#define SEQ     2048
#define BZ      4
#define N_HEADS 32
#define D_HEAD  64
#define CHUNK   128
#define NCHUNK  (SEQ / CHUNK)
#define LN_EPS  1e-5f

// ---------------- bf16 helpers (store as ushort bit patterns) ----------------
typedef __attribute__((ext_vector_type(16))) __bf16 v16bf;
typedef __attribute__((ext_vector_type(8)))  float  v8f;

struct __attribute__((aligned(16))) u32x4 { unsigned int x, y, z, w; };
struct __attribute__((aligned(32))) u32x8 { u32x4 lo, hi; };

static __device__ __forceinline__ unsigned short f2bf(float f) {
    unsigned int u = __builtin_bit_cast(unsigned int, f);
    unsigned int lsb = (u >> 16) & 1u;
    u += 0x7fffu + lsb;                       // round-to-nearest-even
    return (unsigned short)(u >> 16);
}
static __device__ __forceinline__ float bf2f(unsigned short h) {
    unsigned int u = ((unsigned int)h) << 16;
    return __builtin_bit_cast(float, u);
}

// A fragment (16x32 bf16, M x K), source LDS row-major [M][ld]:
// lanes 0-15: M=lane, VGPR0-3 -> K 0..7, VGPR4-7 -> K 16..23
// lanes16-31: M=lane-16, VGPR0-3 -> K 8..15, VGPR4-7 -> K 24..31
static __device__ __forceinline__ v16bf load_frag_A(const unsigned short* base, int ld,
                                                    int m0, int k0, int lane) {
    int hl = lane >> 4, m = lane & 15;
    const unsigned short* row = base + (m0 + m) * ld + k0 + hl * 8;
    u32x4 lo = *reinterpret_cast<const u32x4*>(row);
    u32x4 hi = *reinterpret_cast<const u32x4*>(row + 16);
    return __builtin_bit_cast(v16bf, u32x8{lo, hi});
}
// B fragment (32x16 bf16, K x N), source LDS column-per-lane [N][ld] (K contiguous):
// lanes 0-15: N=lane, K 0..7,16..23 ; lanes 16-31: N=lane-16, K 8..15,24..31
static __device__ __forceinline__ v16bf load_frag_B(const unsigned short* base, int ld,
                                                    int n0, int k0, int lane) {
    int hl = lane >> 4, n = lane & 15;
    const unsigned short* row = base + (n0 + n) * ld + k0 + hl * 8;
    u32x4 lo = *reinterpret_cast<const u32x4*>(row);
    u32x4 hi = *reinterpret_cast<const u32x4*>(row + 16);
    return __builtin_bit_cast(v16bf, u32x8{lo, hi});
}
static __device__ __forceinline__ v8f wmma_bf16(v16bf a, v16bf b, v8f c) {
    return __builtin_amdgcn_wmma_f32_16x16x32_bf16(false, a, false, b, (short)0, c,
                                                   false, false);
}

// ---------------- async global->LDS (CDNA5), guarded ----------------
#if __has_builtin(__builtin_amdgcn_global_load_async_to_lds_b128)
#define HAVE_ASYNC 1
typedef int vsi4 __attribute__((vector_size(16)));
static __device__ __forceinline__ void async_cp16(void* lds_dst, const void* gsrc) {
    __builtin_amdgcn_global_load_async_to_lds_b128((vsi4*)gsrc, (vsi4*)lds_dst, 0, 0);
}
static __device__ __forceinline__ void wait_async_le8() {
#if __has_builtin(__builtin_amdgcn_s_wait_asynccnt)
    __builtin_amdgcn_s_wait_asynccnt(8);
#else
    asm volatile("s_wait_asynccnt 0x8" ::: "memory");
#endif
}
static __device__ __forceinline__ void wait_async_le0() {
#if __has_builtin(__builtin_amdgcn_s_wait_asynccnt)
    __builtin_amdgcn_s_wait_asynccnt(0);
#else
    asm volatile("s_wait_asynccnt 0x0" ::: "memory");
#endif
}
#else
#define HAVE_ASYNC 0
#endif

// ---------------- Kernel 0: weight fp32 [K][N] -> bf16 transposed [N][K] ----------------
__global__ __launch_bounds__(256) void wconv_kernel(const float* __restrict__ W,
                                                    unsigned short* __restrict__ WT) {
    __shared__ unsigned short t[64 * 72];
    const int tid = threadIdx.x;
    const int n0 = blockIdx.x * 64;
    const int k0 = blockIdx.y * 64;
    {
        int kk = tid >> 2;               // 0..63
        int nn = (tid & 3) * 16;         // 0,16,32,48
        const float* row = W + (size_t)(k0 + kk) * D_MODEL + n0 + nn;
        #pragma unroll
        for (int j = 0; j < 16; ++j) t[(nn + j) * 72 + kk] = f2bf(row[j]);
    }
    __syncthreads();
    {
        int rr = tid >> 2;               // n within tile
        int cc = (tid & 3) * 16;         // k within tile
        unsigned short* out = WT + (size_t)(n0 + rr) * D_MODEL + k0 + cc;
        const u32x4* src = reinterpret_cast<const u32x4*>(t + rr * 72 + cc);
        *reinterpret_cast<u32x4*>(out) = src[0];
        *reinterpret_cast<u32x4*>(out + 8) = src[1];
    }
}

// ---------------- Kernel 1: LayerNorm + token-shift mix -> rx/kx/vx bf16 ----------------
__global__ __launch_bounds__(256) void ln_mix_kernel(const float* __restrict__ x,
                              const float* __restrict__ g, const float* __restrict__ bia,
                              const float* __restrict__ tmr, const float* __restrict__ tmk,
                              const float* __restrict__ tmv,
                              unsigned short* __restrict__ rx,
                              unsigned short* __restrict__ kx,
                              unsigned short* __restrict__ vx,
                              float* __restrict__ xn_last) {
    const int row = blockIdx.x;             // 0..BZ*SEQ-1
    const int tid = threadIdx.x;            // 256
    const int lane = tid & 31, wid = tid >> 5;
    const int tin = row & (SEQ - 1);
    const bool hasprev = (tin != 0);
    const float* xr = x + (size_t)row * D_MODEL;
    const float* xp = xr - D_MODEL;

    const int c0 = tid * 8;
    float vc[8], vp[8];
    {
        const float4* p0 = reinterpret_cast<const float4*>(xr + c0);
        float4 a = p0[0], b4 = p0[1];
        vc[0]=a.x; vc[1]=a.y; vc[2]=a.z; vc[3]=a.w; vc[4]=b4.x; vc[5]=b4.y; vc[6]=b4.z; vc[7]=b4.w;
    }
    if (hasprev) {
        const float4* p1 = reinterpret_cast<const float4*>(xp + c0);
        float4 a = p1[0], b4 = p1[1];
        vp[0]=a.x; vp[1]=a.y; vp[2]=a.z; vp[3]=a.w; vp[4]=b4.x; vp[5]=b4.y; vp[6]=b4.z; vp[7]=b4.w;
    } else {
        #pragma unroll
        for (int e = 0; e < 8; ++e) vp[e] = 0.f;
    }
    float p[4] = {0.f, 0.f, 0.f, 0.f};
    #pragma unroll
    for (int e = 0; e < 8; ++e) {
        p[0] += vc[e]; p[1] += vc[e] * vc[e];
        p[2] += vp[e]; p[3] += vp[e] * vp[e];
    }
    __shared__ float red[8][4];
    __shared__ float tot[4];
    #pragma unroll
    for (int off = 16; off > 0; off >>= 1) {
        #pragma unroll
        for (int i = 0; i < 4; ++i) p[i] += __shfl_down(p[i], off, 32);
    }
    if (lane == 0) { red[wid][0]=p[0]; red[wid][1]=p[1]; red[wid][2]=p[2]; red[wid][3]=p[3]; }
    __syncthreads();
    if (tid == 0) {
        #pragma unroll
        for (int i = 0; i < 4; ++i) {
            float s = 0.f;
            for (int w = 0; w < 8; ++w) s += red[w][i];
            tot[i] = s;
        }
    }
    __syncthreads();
    const float inv = 1.0f / (float)D_MODEL;
    float mu_t = tot[0] * inv, var_t = tot[1] * inv - mu_t * mu_t;
    float rs_t = rsqrtf(var_t + LN_EPS);
    float mu_p = tot[2] * inv, var_p = tot[3] * inv - mu_p * mu_p;
    float rs_p = rsqrtf(var_p + LN_EPS);

    const size_t obase = (size_t)row * D_MODEL + c0;
    #pragma unroll
    for (int e = 0; e < 8; ++e) {
        int c = c0 + e;
        float gn = g[c], bn = bia[c];
        float xnt = (vc[e] - mu_t) * rs_t * gn + bn;
        float xnp = hasprev ? ((vp[e] - mu_p) * rs_p * gn + bn) : 0.f;
        float mr = tmr[c], mk = tmk[c], mv = tmv[c];
        rx[obase + e] = f2bf(xnt * mr + (1.f - mr) * xnp);
        kx[obase + e] = f2bf(xnt * mk + (1.f - mk) * xnp);
        vx[obase + e] = f2bf(xnt * mv + (1.f - mv) * xnp);
        if (tin == SEQ - 1) xn_last[(row >> 11) * D_MODEL + c] = xnt;
    }
}

// ---------------- Kernel 2: bf16 WMMA GEMM  Y = X(bf16) @ WT(bf16,[N][K])^T ----------------
// block tile 128x128, 8 waves (2x4), K-step 64, double-buffered async LDS pipeline.
#define GLD 72                  // 64 + 8 pad (halfwords)
#define GSTRIDE (128 * GLD)     // one buffer of one matrix
__global__ __launch_bounds__(256) void gemm_kernel(const unsigned short* __restrict__ X,
                            const unsigned short* __restrict__ WT,
                            void* __restrict__ Yv,
                            const float* __restrict__ resid, int mode) {
    extern __shared__ char gsm[];
    unsigned short* As = (unsigned short*)gsm;        // [2][128][GLD]
    unsigned short* Bs = As + 2 * GSTRIDE;            // [2][128][GLD]
    const int tid = threadIdx.x;
    const int lane = tid & 31, wid = tid >> 5;
    const int wm = wid >> 2, wn = wid & 3;            // 2 x 4 wave grid
    const int m_blk = blockIdx.x * 128;
    const int n_blk = blockIdx.y * 128;
    const int K = D_MODEL, N = D_MODEL;
    const int NKT = K / 64;

    const v8f vzero = {0.f, 0.f, 0.f, 0.f, 0.f, 0.f, 0.f, 0.f};
    v8f acc[4][2];
    #pragma unroll
    for (int i = 0; i < 4; ++i)
        #pragma unroll
        for (int j = 0; j < 2; ++j) acc[i][j] = vzero;

    // staging: each thread owns row r, 32 halfwords starting at ch (4 x b128)
    const int r  = tid & 127;
    const int ch = (tid >> 7) * 32;
    const unsigned short* xrow = X  + (size_t)(m_blk + r) * K + ch;
    const unsigned short* wrow = WT + (size_t)(n_blk + r) * K + ch;
    unsigned short* aldsr = As + r * GLD + ch;
    unsigned short* bldsr = Bs + r * GLD + ch;

#if HAVE_ASYNC
    auto stage = [&](int buf, int k0) {
        #pragma unroll
        for (int q = 0; q < 4; ++q) {
            async_cp16(aldsr + buf * GSTRIDE + q * 8, xrow + k0 + q * 8);
            async_cp16(bldsr + buf * GSTRIDE + q * 8, wrow + k0 + q * 8);
        }
    };
    stage(0, 0);
    for (int kt = 0; kt < NKT; ++kt) {
        const int cur = kt & 1;
        const bool more = (kt + 1 < NKT);
        if (more) stage(cur ^ 1, (kt + 1) * 64);
        if (more) wait_async_le8(); else wait_async_le0();
        __syncthreads();
        const unsigned short* Ab = As + cur * GSTRIDE;
        const unsigned short* Bb = Bs + cur * GSTRIDE;
        #pragma unroll
        for (int kk = 0; kk < 64; kk += 32) {
            v16bf bf[2];
            #pragma unroll
            for (int j = 0; j < 2; ++j) bf[j] = load_frag_B(Bb, GLD, wn * 32 + j * 16, kk, lane);
            #pragma unroll
            for (int i = 0; i < 4; ++i) {
                v16bf af = load_frag_A(Ab, GLD, wm * 64 + i * 16, kk, lane);
                #pragma unroll
                for (int j = 0; j < 2; ++j) acc[i][j] = wmma_bf16(af, bf[j], acc[i][j]);
            }
        }
        __syncthreads();
    }
#else
    for (int kt = 0; kt < NKT; ++kt) {
        const int k0 = kt * 64;
        #pragma unroll
        for (int q = 0; q < 4; ++q) {
            u32x4 d = *reinterpret_cast<const u32x4*>(xrow + k0 + q * 8);
            *reinterpret_cast<u32x4*>(aldsr + q * 8) = d;
            u32x4 e = *reinterpret_cast<const u32x4*>(wrow + k0 + q * 8);
            *reinterpret_cast<u32x4*>(bldsr + q * 8) = e;
        }
        if (kt + 1 < NKT) {
            __builtin_prefetch(xrow + k0 + 64, 0, 1);
            __builtin_prefetch(wrow + k0 + 64, 0, 1);
        }
        __syncthreads();
        #pragma unroll
        for (int kk = 0; kk < 64; kk += 32) {
            v16bf bf[2];
            #pragma unroll
            for (int j = 0; j < 2; ++j) bf[j] = load_frag_B(Bs, GLD, wn * 32 + j * 16, kk, lane);
            #pragma unroll
            for (int i = 0; i < 4; ++i) {
                v16bf af = load_frag_A(As, GLD, wm * 64 + i * 16, kk, lane);
                #pragma unroll
                for (int j = 0; j < 2; ++j) acc[i][j] = wmma_bf16(af, bf[j], acc[i][j]);
            }
        }
        __syncthreads();
    }
#endif

    const int hl = lane >> 4, nl = lane & 15;
    #pragma unroll
    for (int i = 0; i < 4; ++i) {
        #pragma unroll
        for (int j = 0; j < 2; ++j) {
            int mbase = m_blk + wm * 64 + i * 16 + hl * 8;
            int nn = n_blk + wn * 32 + j * 16 + nl;
            if (mode == 0) {
                unsigned short* Y = (unsigned short*)Yv;
                #pragma unroll
                for (int e = 0; e < 8; ++e)
                    Y[(size_t)(mbase + e) * N + nn] = f2bf(acc[i][j][e]);
            } else {
                float* Y = (float*)Yv;
                #pragma unroll
                for (int e = 0; e < 8; ++e) {
                    size_t off = (size_t)(mbase + e) * N + nn;
                    Y[off] = resid[off] + acc[i][j][e];
                }
            }
        }
    }
}

// ---------------- Kernel 3: RWKV chunked attention, one workgroup per (b,h) ----------------
#define LDR 72     // 64+8
#define LDT 136    // 128+8
__global__ __launch_bounds__(256) void attention_kernel(const unsigned short* __restrict__ R,
                                 const unsigned short* __restrict__ Kb,
                                 const unsigned short* __restrict__ Vb,
                                 const float* __restrict__ time_decay,
                                 const float* __restrict__ time_first,
                                 const float* __restrict__ lnx_g,
                                 const float* __restrict__ lnx_b,
                                 unsigned short* __restrict__ attn,
                                 float* __restrict__ state_out) {
    extern __shared__ char smem[];
    unsigned short* r_lds = (unsigned short*)smem;          // [128][LDR]
    unsigned short* k_lds = r_lds + 128 * LDR;              // [128][LDR]   B frags for r@k^T
    unsigned short* kts   = k_lds + 128 * LDR;              // [64][LDT]    k^T * w^(C-1-t)
    unsigned short* vt    = kts + 64 * LDT;                 // [64][LDT]    v^T
    unsigned short* rkw   = vt + 64 * LDT;                  // [128][LDT]   masked r@k^T (bf16)
    unsigned short* sbf   = rkw + 128 * LDT;                // [64][LDR]    state^T bf16 (B frags)
    float* S    = (float*)(sbf + 64 * LDR);                 // [64][64]     fp32 state [d1][d2]
    float* attb = S + 64 * 64;                              // [128][64]    sep+bias fp32

    const int tid = threadIdx.x;
    const int lane = tid & 31, wid = tid >> 5;
    const int hl = lane >> 4, nl = lane & 15;
    const int h = blockIdx.x & (N_HEADS - 1);
    const int b = blockIdx.x >> 5;

    const float ew = __expf(time_decay[h]);     // w = exp(-ew)
    const float u  = __expf(time_first[h]);
    const float wC = __expf(-(float)CHUNK * ew);

    for (int i = tid; i < 64 * 64; i += 256) S[i] = 0.f;
    for (int i = tid; i < 64 * LDR; i += 256) sbf[i] = 0;
    __syncthreads();

    const v8f vzero = {0.f, 0.f, 0.f, 0.f, 0.f, 0.f, 0.f, 0.f};

    for (int c = 0; c < NCHUNK; ++c) {
        const int tg0 = b * SEQ + c * CHUNK;
        // ---- load r,k,v chunk (128 x 64 each) into LDS (k scaled+transposed, v transposed)
        for (int idx = tid; idx < CHUNK * D_HEAD; idx += 256) {
            int t = idx >> 6, d = idx & 63;
            size_t gofs = (size_t)(tg0 + t) * D_MODEL + h * D_HEAD + d;
            unsigned short rv = R[gofs], kv = Kb[gofs], vv = Vb[gofs];
            r_lds[t * LDR + d] = rv;
            k_lds[t * LDR + d] = kv;
            kts[d * LDT + t] = f2bf(bf2f(kv) * __expf(-(float)(CHUNK - 1 - t) * ew));
            vt[d * LDT + t] = vv;
        }
        __syncthreads();

        // ---- rkw = (r @ k^T) * mask, each wave one 16-row strip
        const int m0 = wid * 16;
        {
            v8f racc[8];
            #pragma unroll
            for (int j = 0; j < 8; ++j) racc[j] = vzero;
            #pragma unroll
            for (int kk = 0; kk < 64; kk += 32) {
                v16bf af = load_frag_A(r_lds, LDR, m0, kk, lane);
                #pragma unroll
                for (int j = 0; j < 8; ++j) {
                    v16bf bfb = load_frag_B(k_lds, LDR, j * 16, kk, lane);
                    racc[j] = wmma_bf16(af, bfb, racc[j]);
                }
            }
            #pragma unroll
            for (int j = 0; j < 8; ++j) {
                int t2 = j * 16 + nl;
                #pragma unroll
                for (int e = 0; e < 8; ++e) {
                    int t1 = m0 + e + hl * 8;
                    int dt = t1 - t2;
                    float mval = (dt > 0) ? __expf(-(float)(dt - 1) * ew)
                                          : ((dt == 0) ? u : 0.f);
                    rkw[t1 * LDT + t2] = f2bf(racc[j][e] * mval);
                }
            }
        }
        __syncthreads();

        // ---- att = (r @ state) * w^t  +  rkw @ v
        {
            v8f oacc[4];
            #pragma unroll
            for (int j = 0; j < 4; ++j) oacc[j] = vzero;
            #pragma unroll
            for (int kk = 0; kk < 64; kk += 32) {
                v16bf af = load_frag_A(r_lds, LDR, m0, kk, lane);
                #pragma unroll
                for (int j = 0; j < 4; ++j) {
                    v16bf bfb = load_frag_B(sbf, LDR, j * 16, kk, lane);
                    oacc[j] = wmma_bf16(af, bfb, oacc[j]);
                }
            }
            #pragma unroll
            for (int j = 0; j < 4; ++j) {
                #pragma unroll
                for (int e = 0; e < 8; ++e) {
                    int t1 = m0 + e + hl * 8;
                    oacc[j][e] *= __expf(-(float)t1 * ew);
                }
            }
            #pragma unroll
            for (int kk = 0; kk < 128; kk += 32) {
                v16bf af = load_frag_A(rkw, LDT, m0, kk, lane);
                #pragma unroll
                for (int j = 0; j < 4; ++j) {
                    v16bf bfb = load_frag_B(vt, LDT, j * 16, kk, lane);
                    oacc[j] = wmma_bf16(af, bfb, oacc[j]);
                }
            }
            #pragma unroll
            for (int j = 0; j < 4; ++j)
                #pragma unroll
                for (int e = 0; e < 8; ++e)
                    attb[(m0 + e + hl * 8) * 64 + j * 16 + nl] = oacc[j][e];
        }
        __syncthreads();

        // ---- waves 4-7: state = state*w^C + (k^T*w_k) @ v ; waves 0-3: per-row LN of att
        if (wid >= 4) {
            const int ms = (wid - 4) * 16;
            v8f sacc[4];
            #pragma unroll
            for (int j = 0; j < 4; ++j)
                #pragma unroll
                for (int e = 0; e < 8; ++e)
                    sacc[j][e] = S[(ms + e + hl * 8) * 64 + j * 16 + nl] * wC;
            #pragma unroll
            for (int kk = 0; kk < 128; kk += 32) {
                v16bf af = load_frag_A(kts, LDT, ms, kk, lane);
                #pragma unroll
                for (int j = 0; j < 4; ++j) {
                    v16bf bfb = load_frag_B(vt, LDT, j * 16, kk, lane);
                    sacc[j] = wmma_bf16(af, bfb, sacc[j]);
                }
            }
            #pragma unroll
            for (int j = 0; j < 4; ++j) {
                #pragma unroll
                for (int e = 0; e < 8; ++e) {
                    int m = ms + e + hl * 8, n = j * 16 + nl;
                    float val = sacc[j][e];
                    S[m * 64 + n] = val;
                    sbf[n * LDR + m] = f2bf(val);
                }
            }
        } else if (tid < 128) {
            const int t = tid;
            float mu = 0.f, sq = 0.f;
            #pragma unroll 8
            for (int d = 0; d < 64; ++d) {
                float xx = attb[t * 64 + d];
                mu += xx; sq += xx * xx;
            }
            mu *= (1.f / 64.f);
            float var = sq * (1.f / 64.f) - mu * mu;
            float rs = rsqrtf(var + LN_EPS);
            #pragma unroll 8
            for (int d = 0; d < 64; ++d) {
                float val = (attb[t * 64 + d] - mu) * rs * lnx_g[h * D_HEAD + d]
                            + lnx_b[h * D_HEAD + d];
                attn[(size_t)(tg0 + t) * D_MODEL + h * D_HEAD + d] = f2bf(val);
            }
        }
        __syncthreads();
    }

    // ---- final state out: state[b][h][d1][d2]
    float* so = state_out + (size_t)(b * N_HEADS + h) * D_HEAD * D_HEAD;
    for (int i = tid; i < D_HEAD * D_HEAD; i += 256) so[i] = S[i];
}

// ---------------- host launcher ----------------
extern "C" void kernel_launch(void* const* d_in, const int* in_sizes, int n_in,
                              void* d_out, int out_size, void* d_ws, size_t ws_size,
                              hipStream_t stream) {
    (void)in_sizes; (void)n_in; (void)out_size; (void)ws_size;
    const float* x       = (const float*)d_in[0];
    const float* tmr     = (const float*)d_in[1];
    const float* tmk     = (const float*)d_in[2];
    const float* tmv     = (const float*)d_in[3];
    const float* Wk      = (const float*)d_in[4];
    const float* Wv      = (const float*)d_in[5];
    const float* Wr      = (const float*)d_in[6];
    const float* Wo      = (const float*)d_in[7];
    const float* tdecay  = (const float*)d_in[8];
    const float* tfirst  = (const float*)d_in[9];
    const float* ln1_s   = (const float*)d_in[10];
    const float* ln1_b   = (const float*)d_in[11];
    const float* lnx_s   = (const float*)d_in[12];
    const float* lnx_b   = (const float*)d_in[13];

    // outputs (concatenated flat): [BZ*SEQ*D], [BZ*D], [BZ*H*dh*dh]
    float* out_main = (float*)d_out;
    float* xn_last  = out_main + (size_t)BZ * SEQ * D_MODEL;
    float* state_o  = xn_last + (size_t)BZ * D_MODEL;

    // workspace: six 32 MiB activation panels + four 8 MiB bf16 transposed weights
    const size_t PANEL = (size_t)BZ * SEQ * D_MODEL * sizeof(unsigned short); // 32 MiB
    const size_t WSZ   = (size_t)D_MODEL * D_MODEL * sizeof(unsigned short);  // 8 MiB
    char* ws = (char*)d_ws;
    unsigned short* RX   = (unsigned short*)(ws + 0 * PANEL);
    unsigned short* KX   = (unsigned short*)(ws + 1 * PANEL);
    unsigned short* VX   = (unsigned short*)(ws + 2 * PANEL);
    unsigned short* RB   = (unsigned short*)(ws + 3 * PANEL);
    unsigned short* KB   = (unsigned short*)(ws + 4 * PANEL);
    unsigned short* VB   = (unsigned short*)(ws + 5 * PANEL);
    unsigned short* WTr  = (unsigned short*)(ws + 6 * PANEL + 0 * WSZ);
    unsigned short* WTk  = (unsigned short*)(ws + 6 * PANEL + 1 * WSZ);
    unsigned short* WTv  = (unsigned short*)(ws + 6 * PANEL + 2 * WSZ);
    unsigned short* WTo  = (unsigned short*)(ws + 6 * PANEL + 3 * WSZ);
    unsigned short* ATTN = RX;   // RX consumed before attention writes it

    dim3 wc(D_MODEL / 64, D_MODEL / 64);
    wconv_kernel<<<wc, 256, 0, stream>>>(Wr, WTr);
    wconv_kernel<<<wc, 256, 0, stream>>>(Wk, WTk);
    wconv_kernel<<<wc, 256, 0, stream>>>(Wv, WTv);
    wconv_kernel<<<wc, 256, 0, stream>>>(Wo, WTo);

    ln_mix_kernel<<<BZ * SEQ, 256, 0, stream>>>(x, ln1_s, ln1_b, tmr, tmk, tmv,
                                                RX, KX, VX, xn_last);

    dim3 gg(BZ * SEQ / 128, D_MODEL / 128);
    const size_t gemm_smem = (size_t)4 * GSTRIDE * sizeof(unsigned short); // 2 bufs x (A+B)
    gemm_kernel<<<gg, 256, gemm_smem, stream>>>(RX, WTr, RB, nullptr, 0);
    gemm_kernel<<<gg, 256, gemm_smem, stream>>>(KX, WTk, KB, nullptr, 0);
    gemm_kernel<<<gg, 256, gemm_smem, stream>>>(VX, WTv, VB, nullptr, 0);

    const size_t att_smem =
        (size_t)(128 * LDR + 128 * LDR + 64 * LDT + 64 * LDT + 128 * LDT + 64 * LDR) * 2
        + (size_t)(64 * 64 + 128 * 64) * 4;   // ~161 KB, under 320 KB/WGP
    attention_kernel<<<BZ * N_HEADS, 256, att_smem, stream>>>(
        RB, KB, VB, tdecay, tfirst, lnx_s, lnx_b, ATTN, state_o);

    gemm_kernel<<<gg, 256, gemm_smem, stream>>>(ATTN, WTo, out_main, x, 1);
}
